// CausalSelfAttention_36679020708327
// MI455X (gfx1250) — matrix operand order repaired and checked by reading, not compiled, not run on hardware
//
#include <hip/hip_runtime.h>

// Problem constants (match reference)
#define BB 4
#define TT 2048
#define CCH 1024
#define HH 16
#define DH 64
#define MROWS (BB * TT)   // 8192
#define N_QKV (3 * CCH)   // 3072

// LDS row strides (elements). Multiples of 8 keep ds_load_b128 16B-aligned;
// 72 -> 36-dword row stride, 40 -> 20-dword: 16 consecutive rows map to 16
// distinct banks for the fragment readers.
#define KSTR 72
#define VSTR 40
#define PSTR 40

typedef __attribute__((ext_vector_type(16))) __bf16 v16bf;
typedef __attribute__((ext_vector_type(8)))  float  v8f;
typedef __attribute__((ext_vector_type(4)))  unsigned int u32x4;
typedef __attribute__((ext_vector_type(8)))  unsigned int u32x8;
typedef __attribute__((ext_vector_type(4)))  int          i32x4;

#define WMMA_BF16(a, b, c) \
    __builtin_amdgcn_wmma_f32_16x16x32_bf16(false, (a), false, (b), (short)0, (c), false, false)

#if __has_builtin(__builtin_amdgcn_global_load_async_to_lds_b128)
#define HAVE_ASYNC 1
#else
#define HAVE_ASYNC 0
#endif

static __device__ __forceinline__ unsigned short f32_to_bf16(float f) {
    unsigned int u = __float_as_uint(f);
    u += 0x7fffu + ((u >> 16) & 1u);   // round to nearest even
    return (unsigned short)(u >> 16);
}

static __device__ __forceinline__ v16bf fuse16(u32x4 lo, u32x4 hi) {
    u32x8 c;
    c[0] = lo[0]; c[1] = lo[1]; c[2] = lo[2]; c[3] = lo[3];
    c[4] = hi[0]; c[5] = hi[1]; c[6] = hi[2]; c[7] = hi[3];
    return __builtin_bit_cast(v16bf, c);
}

// A-operand fragment (16x32 bf16). p = row_ptr + kbase + ((lane&16)?8:0).
static __device__ __forceinline__ v16bf load_fragA(const unsigned short* p) {
    return fuse16(*(const u32x4*)p, *(const u32x4*)(p + 16));
}
// B-operand fragment (32x16 bf16). p = col_ptr + kbase + ((lane&16)?8:0).
static __device__ __forceinline__ v16bf load_fragB(const unsigned short* p) {
    return fuse16(*(const u32x4*)p, *(const u32x4*)(p + 16));
}

// 16-byte global -> LDS copy: async DMA (tracked by ASYNCcnt) if the toolchain
// exposes it, else synchronous load + ds_store_b128.
static __device__ __forceinline__ void cp16(const unsigned short* g, unsigned short* l) {
#if HAVE_ASYNC
    __builtin_amdgcn_global_load_async_to_lds_b128(
        (__attribute__((address_space(1))) i32x4*)(unsigned short*)g,
        (__attribute__((address_space(3))) i32x4*)l,
        0, 0);
#else
    *(u32x4*)l = *(const u32x4*)g;
#endif
}

// ---------------- cast / transpose helpers ----------------

__global__ __launch_bounds__(256) void cvt_f32_bf16(const float* __restrict__ in,
                                                    unsigned short* __restrict__ out,
                                                    size_t n) {
    size_t i = (size_t)blockIdx.x * 256 + threadIdx.x;
    if (i < n) out[i] = f32_to_bf16(in[i]);
}

// in [R x Ccols] f32 row-major -> out [Ccols x R] bf16 (transposed)
__global__ __launch_bounds__(256) void transpose_cvt(const float* __restrict__ in,
                                                     unsigned short* __restrict__ out,
                                                     int R, int Ccols) {
    int c = blockIdx.x * 32 + (threadIdx.x & 31);
    int r = blockIdx.y * 8 + (threadIdx.x >> 5);
    if (c < Ccols && r < R)
        out[(size_t)c * R + r] = f32_to_bf16(in[(size_t)r * Ccols + c]);
}

// ---------------- WMMA GEMM: out[M,N] = A[M,K] * Bt[N,K]^T + bias ----------------
// 64x64 tile per wave (16 WMMAs / K=32 step), 2x2 waves -> 128x128 block tile,
// register-double-buffered fragments so next-K loads overlap current WMMAs.
// MODE 0: QKV epilogue -> bf16 q[B,H,T,D], k[B,H,T,D], vT[B,H,D,T]
//         (q/k/v selector, batch, head, channel are hoisted: 128 | 1024, 2048)
// MODE 1: proj epilogue -> f32 row-major [M, CCH]

template <int MODE>
__global__ __launch_bounds__(128) void gemm_bf16_k(const unsigned short* __restrict__ A,
                                                   const unsigned short* __restrict__ Bt,
                                                   const float* __restrict__ bias,
                                                   unsigned short* __restrict__ qb,
                                                   unsigned short* __restrict__ kb,
                                                   unsigned short* __restrict__ vb,
                                                   float* __restrict__ outf,
                                                   int Kdim) {
    const int lane = threadIdx.x & 31;
    const int wave = threadIdx.x >> 5;
    const int l16  = lane & 15;
    const int aoff = (lane & 16) ? 8 : 0;
    const int boff = (lane & 16) ? 8 : 0;
    const int hi8  = (lane & 16) ? 8 : 0;
    const int mbase = blockIdx.x * 128 + (wave >> 1) * 64;
    const int nbase = blockIdx.y * 128 + (wave & 1) * 64;

    const unsigned short* ap[4];
    const unsigned short* bp[4];
#pragma unroll
    for (int t = 0; t < 4; ++t) {
        ap[t] = A  + (size_t)(mbase + t * 16 + l16) * Kdim + aoff;
        bp[t] = Bt + (size_t)(nbase + t * 16 + l16) * Kdim + boff;
    }

    v8f acc[4][4] = {};
    v16bf af[4], bf[4];
#pragma unroll
    for (int t = 0; t < 4; ++t) { af[t] = load_fragA(ap[t]); bf[t] = load_fragB(bp[t]); }

    for (int kk = 0; kk < Kdim; kk += 32) {
        v16bf an[4], bn[4];
        const bool more = (kk + 32) < Kdim;
        if (more) {
#pragma unroll
            for (int t = 0; t < 4; ++t) {
                an[t] = load_fragA(ap[t] + kk + 32);
                bn[t] = load_fragB(bp[t] + kk + 32);
            }
        }
#pragma unroll
        for (int i = 0; i < 4; ++i)
#pragma unroll
            for (int j = 0; j < 4; ++j)
                acc[i][j] = WMMA_BF16(af[i], bf[j], acc[i][j]);
        if (more) {
#pragma unroll
            for (int t = 0; t < 4; ++t) { af[t] = an[t]; bf[t] = bn[t]; }
        }
    }

    float bv[4];
#pragma unroll
    for (int t = 0; t < 4; ++t) bv[t] = bias[nbase + t * 16 + l16];

    if (MODE == 0) {
        const int which = nbase >> 10;    // block-uniform: 0=q 1=k 2=v
        const int ccb   = nbase & 1023;   // base channel within q/k/v
        const int bidx  = mbase >> 11;    // batch index (block-uniform)
        const int tbase = mbase & 2047;   // t of tile row 0
        unsigned short* dst = (which == 0) ? qb : (which == 1) ? kb : vb;
#pragma unroll
        for (int j = 0; j < 4; ++j) {
            const int cc = ccb + j * 16;
            const size_t bhj = (size_t)(bidx * HH + (cc >> 6));
            const int dj = (cc & 63) + l16;
            if (which < 2) {
                // [bh][t][d]: per-element b16 stores, stride DH in t
                unsigned short* base = dst + (bhj * TT + tbase) * DH + dj;
#pragma unroll
                for (int i = 0; i < 4; ++i)
#pragma unroll
                    for (int r = 0; r < 8; ++r)
                        base[(size_t)(i * 16 + hi8 + r) * DH] =
                            f32_to_bf16(acc[i][j][r] + bv[j]);
            } else {
                // vT: [bh][d][t]: the 8 rows of a tile are contiguous in t ->
                // pack to one aligned 16B store per tile.
                unsigned short* base = dst + (bhj * DH + dj) * TT + tbase;
#pragma unroll
                for (int i = 0; i < 4; ++i) {
                    u32x4 pk;
#pragma unroll
                    for (int r2 = 0; r2 < 4; ++r2) {
                        unsigned lo = f32_to_bf16(acc[i][j][2 * r2]     + bv[j]);
                        unsigned hi = f32_to_bf16(acc[i][j][2 * r2 + 1] + bv[j]);
                        pk[r2] = lo | (hi << 16);
                    }
                    *(u32x4*)(base + i * 16 + hi8) = pk;
                }
            }
        }
    } else {
#pragma unroll
        for (int i = 0; i < 4; ++i)
#pragma unroll
            for (int r = 0; r < 8; ++r) {
                float* orow = outf + (size_t)(mbase + i * 16 + hi8 + r) * CCH + nbase + l16;
#pragma unroll
                for (int j = 0; j < 4; ++j)
                    orow[j * 16] = acc[i][j][r] + bv[j];
            }
    }
}

// ---------------- flash attention (causal, online softmax) ----------------
// Block = 4 waves, each wave owns 16 query rows. 32-key tiles of K and V are
// cooperatively staged into double-buffered LDS (async-to-LDS), shared by all
// 4 waves; per-wave P bounces through private LDS to become the A operand of
// P*V.

__global__ __launch_bounds__(128) void flash_attn_k(const unsigned short* __restrict__ qbuf,
                                                    const unsigned short* __restrict__ kbuf,
                                                    const unsigned short* __restrict__ vtbuf,
                                                    unsigned short* __restrict__ ybuf) {
    __shared__ __align__(16) unsigned short ldsK[2][32 * KSTR];  // [key][d]
    __shared__ __align__(16) unsigned short ldsV[2][64 * VSTR];  // [d][key]
    __shared__ __align__(16) unsigned short ldsP[4][16 * PSTR];  // per-wave

    const int tid  = threadIdx.x;
    const int lane = tid & 31;
    const int wave = tid >> 5;
    const int l16  = lane & 15;
    const int hi8  = (lane & 16) ? 8 : 0;
    const int aoff = hi8;
    const int boff = (lane & 16) ? 8 : 0;

    const int bh = blockIdx.y;          // b*H + h
    const int b  = bh >> 4;
    const int h  = bh & 15;
    const int qbase = (blockIdx.x * 4 + wave) * 16;

    const size_t headoff = (size_t)bh * TT * DH;
    const unsigned short* Q  = qbuf + headoff;
    const unsigned short* Km = kbuf + headoff;
    const unsigned short* Vt = vtbuf + headoff;

    const unsigned short* qrow = Q + (size_t)(qbase + l16) * DH + aoff;
    v16bf qa0 = load_fragA(qrow);        // d 0..31
    v16bf qa1 = load_fragA(qrow + 32);   // d 32..63

    v8f acc[4] = {};
    float mrun[8], lrun[8];
#pragma unroll
    for (int r = 0; r < 8; ++r) { mrun[r] = -3.0e38f; lrun[r] = 0.0f; }

    const float sc = 0.125f * 1.44269504088896340736f;  // 1/sqrt(D) * log2(e)
    unsigned short* lp = ldsP[wave];

    const int kend   = qbase + 16;                  // this wave's causal range
    const int ntiles = (blockIdx.x * 64 + 64) >> 5; // block-wide tile count

    // cooperative stage of one 32-key tile: 4 x 16B async copies per thread
    auto stage = [&](int tile, int buf) {
#pragma unroll
        for (int i = 0; i < 2; ++i) {
            int ch   = tid + (i << 7);
            int krow = ch >> 3, kc = (ch & 7) << 3;
            cp16(Km + (size_t)(tile * 32 + krow) * DH + kc,
                 &ldsK[buf][krow * KSTR + kc]);
            int vrow = ch >> 2, vc = (ch & 3) << 3;
            cp16(Vt + (size_t)vrow * TT + tile * 32 + vc,
                 &ldsV[buf][vrow * VSTR + vc]);
        }
    };
    stage(0, 0);

    for (int it = 0; it < ntiles; ++it) {
        const int kt = it << 5;
        const bool more = (it + 1) < ntiles;
        if (more) stage(it + 1, (it + 1) & 1);
#if HAVE_ASYNC
        // 4 prefetch copies may stay in flight; tile `it` is then resident.
        if (more) asm volatile("s_wait_asynccnt 0x4" ::: "memory");
        else      asm volatile("s_wait_asynccnt 0x0" ::: "memory");
#endif
        __syncthreads();

        if (kt < kend) {   // wave-uniform; EXEC stays all-ones for WMMA
            const unsigned short* lk = ldsK[it & 1];
            const unsigned short* lv = ldsV[it & 1];

            const unsigned short* kc0 = lk + (size_t)l16 * KSTR + boff;
            const unsigned short* kc1 = lk + (size_t)(16 + l16) * KSTR + boff;
            v16bf k0a = load_fragB(kc0);
            v16bf k0b = load_fragB(kc0 + 32);
            v16bf k1a = load_fragB(kc1);
            v16bf k1b = load_fragB(kc1 + 32);
            v8f s0 = {}, s1 = {};
            s0 = WMMA_BF16(qa0, k0a, s0);
            s0 = WMMA_BF16(qa1, k0b, s0);
            s1 = WMMA_BF16(qa0, k1a, s1);
            s1 = WMMA_BF16(qa1, k1b, s1);

#pragma unroll
            for (int r = 0; r < 8; ++r) { s0[r] *= sc; s1[r] *= sc; }

            if (kt + 31 > qbase) {  // diagonal tile: causal mask
#pragma unroll
                for (int r = 0; r < 8; ++r) {
                    int rr = qbase + hi8 + r;
                    if (kt + l16 > rr)      s0[r] = -3.0e38f;
                    if (kt + 16 + l16 > rr) s1[r] = -3.0e38f;
                }
            }

            // online softmax: rows 0..7 in lanes 0-15, rows 8..15 in lanes 16-31
            float alpha[8];
#pragma unroll
            for (int r = 0; r < 8; ++r) {
                float mx = fmaxf(s0[r], s1[r]);
#pragma unroll
                for (int d = 1; d < 16; d <<= 1) mx = fmaxf(mx, __shfl_xor(mx, d, 32));
                float mnew = fmaxf(mrun[r], mx);
                float a  = exp2f(mrun[r] - mnew);
                float p0 = exp2f(s0[r] - mnew);
                float p1 = exp2f(s1[r] - mnew);
                s0[r] = p0; s1[r] = p1;
                float rs = p0 + p1;
#pragma unroll
                for (int d = 1; d < 16; d <<= 1) rs += __shfl_xor(rs, d, 32);
                lrun[r] = lrun[r] * a + rs;
                mrun[r] = mnew;
                alpha[r] = a;
            }
#pragma unroll
            for (int r = 0; r < 8; ++r) {
                acc[0][r] *= alpha[r]; acc[1][r] *= alpha[r];
                acc[2][r] *= alpha[r]; acc[3][r] *= alpha[r];
            }

            // C-layout P -> bf16 LDS -> A fragment (per-wave region, DS in-order)
#pragma unroll
            for (int r = 0; r < 8; ++r) {
                int m = hi8 + r;
                lp[m * PSTR + l16]      = f32_to_bf16(s0[r]);
                lp[m * PSTR + 16 + l16] = f32_to_bf16(s1[r]);
            }
            v16bf pa = load_fragA(lp + (size_t)l16 * PSTR + aoff);

            // acc += P * V, V fragments from LDS (column n = d, K = key)
#pragma unroll
            for (int nt = 0; nt < 4; ++nt) {
                v16bf vf = load_fragB(lv + (size_t)(nt * 16 + l16) * VSTR + boff);
                acc[nt] = WMMA_BF16(pa, vf, acc[nt]);
            }
        }
        __syncthreads();   // protect buffer parity reuse by next prefetch
    }

    // y[b, t, h*64 + d] = acc / l   (bf16 for the proj GEMM)
#pragma unroll
    for (int r = 0; r < 8; ++r) {
        float inv = 1.0f / lrun[r];
        size_t row = (size_t)b * TT + qbase + hi8 + r;
        unsigned short* yp = ybuf + row * CCH + h * DH + l16;
        yp[0]  = f32_to_bf16(acc[0][r] * inv);
        yp[16] = f32_to_bf16(acc[1][r] * inv);
        yp[32] = f32_to_bf16(acc[2][r] * inv);
        yp[48] = f32_to_bf16(acc[3][r] * inv);
    }
}

// ---------------- launch ----------------

extern "C" void kernel_launch(void* const* d_in, const int* in_sizes, int n_in,
                              void* d_out, int out_size, void* d_ws, size_t ws_size,
                              hipStream_t stream) {
    (void)in_sizes; (void)n_in; (void)out_size; (void)ws_size;
    const float* x      = (const float*)d_in[0];
    const float* w_qkv  = (const float*)d_in[1];
    const float* b_qkv  = (const float*)d_in[2];
    const float* w_proj = (const float*)d_in[3];
    const float* b_proj = (const float*)d_in[4];
    float* out = (float*)d_out;

    char* ws = (char*)d_ws;
    size_t off = 0;
    auto take = [&](size_t elems) -> unsigned short* {
        unsigned short* p = (unsigned short*)(ws + off);
        off += ((elems * 2) + 255) & ~(size_t)255;
        return p;
    };
    unsigned short* xb   = take((size_t)MROWS * CCH);        // x bf16        16 MB
    unsigned short* wqt  = take((size_t)N_QKV * CCH);        // w_qkv^T bf16   6 MB
    unsigned short* wpt  = take((size_t)CCH * CCH);          // w_proj^T bf16  2 MB
    unsigned short* qbuf = take((size_t)BB * HH * TT * DH);  // 16 MB
    unsigned short* kbuf = take((size_t)BB * HH * TT * DH);  // 16 MB
    unsigned short* vbuf = take((size_t)BB * HH * TT * DH);  // 16 MB (transposed)
    unsigned short* ybuf = take((size_t)MROWS * CCH);        // 16 MB

    {
        size_t n = (size_t)MROWS * CCH;
        cvt_f32_bf16<<<dim3((unsigned)(n / 256)), dim3(256), 0, stream>>>(x, xb, n);
    }
    transpose_cvt<<<dim3(N_QKV / 32, CCH / 8), dim3(256), 0, stream>>>(w_qkv, wqt, CCH, N_QKV);
    transpose_cvt<<<dim3(CCH / 32, CCH / 8), dim3(256), 0, stream>>>(w_proj, wpt, CCH, CCH);

    gemm_bf16_k<0><<<dim3(MROWS / 128, N_QKV / 128), dim3(128), 0, stream>>>(
        xb, wqt, b_qkv, qbuf, kbuf, vbuf, nullptr, CCH);

    flash_attn_k<<<dim3(TT / 64, BB * HH), dim3(128), 0, stream>>>(qbuf, kbuf, vbuf, ybuf);

    gemm_bf16_k<1><<<dim3(MROWS / 128, CCH / 128), dim3(128), 0, stream>>>(
        ybuf, wpt, b_proj, nullptr, nullptr, nullptr, out, CCH);
}
